// DeepGNNAutoEncoder_88313117541118
// MI455X (gfx1250) — matrix operand order repaired and checked by reading, not compiled, not run on hardware
//
#include <hip/hip_runtime.h>
#include <hip/hip_bf16.h>

typedef __attribute__((ext_vector_type(16))) __bf16 v16bf;
typedef __attribute__((ext_vector_type(8)))  float  v8f;

// ---------------- elementwise / graph-prep kernels ----------------

__global__ void fill_f32(float* __restrict__ p, float v, long n) {
    long i = (long)blockIdx.x * blockDim.x + threadIdx.x;
    if (i < n) p[i] = v;
}

__global__ void deg_accum(const int* __restrict__ dst, float* __restrict__ deg, int E) {
    int e = blockIdx.x * blockDim.x + threadIdx.x;
    if (e < E) atomicAdd(&deg[dst[e]], 1.0f);
}

__global__ void rsqrt_inplace(float* __restrict__ d, int n) {
    int i = blockIdx.x * blockDim.x + threadIdx.x;
    if (i < n) d[i] = rsqrtf(d[i]);
}

__global__ void edge_norm(const int* __restrict__ src, const int* __restrict__ dst,
                          const float* __restrict__ dinv, float* __restrict__ nrm, int E) {
    int e = blockIdx.x * blockDim.x + threadIdx.x;
    if (e < E) nrm[e] = dinv[src[e]] * dinv[dst[e]];
}

// gather h[src]*norm, scatter-add to out[dst]; one float4 channel-group per thread
__global__ void aggregate(const float* __restrict__ h, const int* __restrict__ src,
                          const int* __restrict__ dst, const float* __restrict__ nrm,
                          float* __restrict__ out, long total, int cshift, int cmask, int outC) {
    long g = (long)blockIdx.x * blockDim.x + threadIdx.x;
    if (g >= total) return;
    int e  = (int)(g >> cshift);
    int c4 = (int)((g & (long)cmask) << 2);
    int s = src[e], d = dst[e];
    float w = nrm[e];
    const float4 v = *(const float4*)(h + (long)s * outC + c4);
    float* o = out + (long)d * outC + c4;
    atomicAdd(o + 0, v.x * w);
    atomicAdd(o + 1, v.y * w);
    atomicAdd(o + 2, v.z * w);
    atomicAdd(o + 3, v.w * w);
}

// out[i,c] = (out[i,c] + h[i,c]*dinv[i]^2 + b[c]), optional ReLU (self-loop folded in)
__global__ void finalize(float* __restrict__ out, const float* __restrict__ h,
                         const float* __restrict__ dinv, const float* __restrict__ bias,
                         long total, int cshift, int cmask, int relu) {
    long g = (long)blockIdx.x * blockDim.x + threadIdx.x;
    if (g >= total) return;
    int i = (int)(g >> cshift);
    int c = (int)(g & (long)cmask);
    float di = dinv[i];
    float v = out[g] + h[g] * di * di + bias[c];
    out[g] = relu ? fmaxf(v, 0.0f) : v;
}

// ---------------- WMMA bf16 GEMM:  out[N,OUTC] = act[N,INC] @ W[OUTC,INC]^T ----------------
// One wave32 computes a 16-row strip across all OUTC columns; K-loop in steps of 32.
// A/B lane layouts per CDNA5 ISA 7.12.2 (16-bit 16x32 A, 32x16 B); f32 accumulate.

template<int INC, int OUTC>
__global__ __launch_bounds__(256)
void gemm_wmma(const float* __restrict__ act, const float* __restrict__ Wt,
               float* __restrict__ out, int N) {
    constexpr int NT = OUTC / 16;   // N tiles
    constexpr int KT = INC / 32;    // K steps
    const int lane = threadIdx.x & 31;
    const int wave = threadIdx.x >> 5;
    const int half = lane >> 4;     // 0: lanes 0-15, 1: lanes 16-31
    const int l15  = lane & 15;
    const int rowBase = (blockIdx.x * 8 + wave) * 16;
    if (rowBase >= N) return;       // wave-uniform: EXEC stays all-ones for WMMA

    const v8f vzero = {0.f,0.f,0.f,0.f,0.f,0.f,0.f,0.f};
    v8f acc[NT];
#pragma unroll
    for (int t = 0; t < NT; ++t) acc[t] = vzero;

    const int   m    = rowBase + l15;          // A row for this lane (N % 16 == 0)
    const float* arow = act + (long)m * INC;

#pragma unroll
    for (int ks = 0; ks < KT; ++ks) {
        const int kBase = ks * 32;
        // A tile 16x32: lanes 0-15 hold K in [0,8)u[16,24); lanes 16-31 hold [8,16)u[24,32)
        const float4* ap0 = (const float4*)(arow + kBase + half * 8);
        const float4* ap1 = (const float4*)(arow + kBase + 16 + half * 8);
        float4 f0 = ap0[0], f1 = ap0[1];
        float4 f2 = ap1[0], f3 = ap1[1];
        v16bf a;
        a[0]=(__bf16)f0.x;  a[1]=(__bf16)f0.y;  a[2]=(__bf16)f0.z;  a[3]=(__bf16)f0.w;
        a[4]=(__bf16)f1.x;  a[5]=(__bf16)f1.y;  a[6]=(__bf16)f1.z;  a[7]=(__bf16)f1.w;
        a[8]=(__bf16)f2.x;  a[9]=(__bf16)f2.y;  a[10]=(__bf16)f2.z; a[11]=(__bf16)f2.w;
        a[12]=(__bf16)f3.x; a[13]=(__bf16)f3.y; a[14]=(__bf16)f3.z; a[15]=(__bf16)f3.w;

#pragma unroll
        for (int nt = 0; nt < NT; ++nt) {
            // B tile 32x16: lane holds column N=l15, K in [0,8)u[16,24) for lanes 0-15, [8,16)u[24,32) for lanes 16-31
            const int n = nt * 16 + l15;
            const float* wrow = Wt + (long)n * INC + kBase + half * 8;
            float4 w0 = ((const float4*)wrow)[0];
            float4 w1 = ((const float4*)wrow)[1];
            float4 w2 = ((const float4*)wrow)[4];
            float4 w3 = ((const float4*)wrow)[5];
            v16bf b;
            b[0]=(__bf16)w0.x;  b[1]=(__bf16)w0.y;  b[2]=(__bf16)w0.z;  b[3]=(__bf16)w0.w;
            b[4]=(__bf16)w1.x;  b[5]=(__bf16)w1.y;  b[6]=(__bf16)w1.z;  b[7]=(__bf16)w1.w;
            b[8]=(__bf16)w2.x;  b[9]=(__bf16)w2.y;  b[10]=(__bf16)w2.z; b[11]=(__bf16)w2.w;
            b[12]=(__bf16)w3.x; b[13]=(__bf16)w3.y; b[14]=(__bf16)w3.z; b[15]=(__bf16)w3.w;

            acc[nt] = __builtin_amdgcn_wmma_f32_16x16x32_bf16(
                false, a, false, b, (short)0, acc[nt], false, false);
        }
    }

    // C/D layout: VGPR v -> row rowBase + v + 8*half, col nt*16 + l15
#pragma unroll
    for (int nt = 0; nt < NT; ++nt) {
#pragma unroll
        for (int v = 0; v < 8; ++v) {
            int row = rowBase + half * 8 + v;
            out[(long)row * OUTC + nt * 16 + l15] = acc[nt][v];
        }
    }
}

// ---------------- launch ----------------

extern "C" void kernel_launch(void* const* d_in, const int* in_sizes, int n_in,
                              void* d_out, int out_size, void* d_ws, size_t ws_size,
                              hipStream_t stream) {
    const float* x  = (const float*)d_in[0];
    const int*   ei = (const int*)d_in[1];   // int32 edge_index [2,E] (JAX x64 off)
    const float* W1 = (const float*)d_in[2];
    const float* b1 = (const float*)d_in[3];
    const float* W2 = (const float*)d_in[4];
    const float* b2 = (const float*)d_in[5];
    const float* W3 = (const float*)d_in[6];
    const float* b3 = (const float*)d_in[7];
    const float* W4 = (const float*)d_in[8];
    const float* b4 = (const float*)d_in[9];

    const int N = in_sizes[0] / 128;     // 100000
    const int E = in_sizes[1] / 2;       // 1600000
    const int* src = ei;
    const int* dst = ei + E;
    float* out = (float*)d_out;

    // workspace carve-up (256B aligned)
    char*  ws  = (char*)d_ws;
    size_t off = 0;
    auto carve = [&](size_t bytes) -> float* {
        float* p = (float*)(ws + off);
        off = (off + bytes + 255) & ~(size_t)255;
        return p;
    };
    float* dinv = carve((size_t)N * 4);          // deg, then rsqrt in place
    float* nrm  = carve((size_t)E * 4);
    float* hbuf = carve((size_t)N * 128 * 4);
    float* bufA = carve((size_t)N * 128 * 4);
    float* bufB = carve((size_t)N * 64 * 4);

    const int TB = 256;
    auto blocks = [&](long n) -> unsigned { return (unsigned)((n + TB - 1) / TB); };

    // graph prep: deg (incl. self-loop) -> dinv -> per-edge norm (reused by all 4 layers)
    fill_f32<<<blocks(N), TB, 0, stream>>>(dinv, 1.0f, N);
    deg_accum<<<blocks(E), TB, 0, stream>>>(dst, dinv, E);
    rsqrt_inplace<<<blocks(N), TB, 0, stream>>>(dinv, N);
    edge_norm<<<blocks(E), TB, 0, stream>>>(src, dst, dinv, nrm, E);

    const unsigned gB = (unsigned)((N + 127) / 128);   // 8 waves x 16 rows per block
    const long t128 = (long)N * 128, t64 = (long)N * 64;
    const long e128 = (long)E * 32,  e64 = (long)E * 16;

    // Layer 1: 128 -> 128, ReLU, out = bufA
    gemm_wmma<128,128><<<gB, 256, 0, stream>>>(x, W1, hbuf, N);
    fill_f32<<<blocks(t128), TB, 0, stream>>>(bufA, 0.0f, t128);
    aggregate<<<blocks(e128), TB, 0, stream>>>(hbuf, src, dst, nrm, bufA, e128, 5, 31, 128);
    finalize<<<blocks(t128), TB, 0, stream>>>(bufA, hbuf, dinv, b1, t128, 7, 127, 1);

    // Layer 2: 128 -> 64, out = bufB
    gemm_wmma<128,64><<<gB, 256, 0, stream>>>(bufA, W2, hbuf, N);
    fill_f32<<<blocks(t64), TB, 0, stream>>>(bufB, 0.0f, t64);
    aggregate<<<blocks(e64), TB, 0, stream>>>(hbuf, src, dst, nrm, bufB, e64, 4, 15, 64);
    finalize<<<blocks(t64), TB, 0, stream>>>(bufB, hbuf, dinv, b2, t64, 6, 63, 0);

    // Layer 3: 64 -> 128, ReLU, out = bufA (free after L2 GEMM consumed it)
    gemm_wmma<64,128><<<gB, 256, 0, stream>>>(bufB, W3, hbuf, N);
    fill_f32<<<blocks(t128), TB, 0, stream>>>(bufA, 0.0f, t128);
    aggregate<<<blocks(e128), TB, 0, stream>>>(hbuf, src, dst, nrm, bufA, e128, 5, 31, 128);
    finalize<<<blocks(t128), TB, 0, stream>>>(bufA, hbuf, dinv, b3, t128, 7, 127, 1);

    // Layer 4: 128 -> 128, out = d_out
    gemm_wmma<128,128><<<gB, 256, 0, stream>>>(bufA, W4, hbuf, N);
    fill_f32<<<blocks(t128), TB, 0, stream>>>(out, 0.0f, t128);
    aggregate<<<blocks(e128), TB, 0, stream>>>(hbuf, src, dst, nrm, out, e128, 5, 31, 128);
    finalize<<<blocks(t128), TB, 0, stream>>>(out, hbuf, dinv, b4, t128, 7, 127, 0);
}
